// LSTMCellModel_55207509623227
// MI455X (gfx1250) — matrix operand order repaired and checked by reading, not compiled, not run on hardware
//
#include <hip/hip_runtime.h>
#include <hip/hip_bf16.h>

typedef __attribute__((ext_vector_type(16))) _Float16 v16h;
typedef __attribute__((ext_vector_type(8)))  _Float16 v8h;
typedef __attribute__((ext_vector_type(8)))  float    v8f;

constexpr int HC    = 200;            // hidden cells
constexpr int BATCH = 32;
constexpr int KPAD  = 224;            // 200 h + [200]=x_t/0 + [201]=1.0(bias) + zero pad
constexpr int NKC   = KPAD / 32;      // 7 K-chunks of 32
constexpr int NNT   = (4 * HC) / 16;  // 50 gate tiles of 16 (gate-interleaved n = 4u+gate)
constexpr int PANEL_HALFS = NNT * NKC * 32 * 16;  // 179200 f16 per weight panel
constexpr int NWAVE = 16;
constexpr int NTHREADS = NWAVE * 32;
constexpr int NTMAX = 4;              // waves 0,1 own 4 tiles; others 3

// Dynamic LDS partition (bytes, all 32B aligned)
constexpr int HPANEL_HALFS = 2 * NKC * 32 * 16;   // 7168 halfs per h buffer
constexpr int H1A_OFF = 0;
constexpr int H2A_OFF = H1A_OFF + HPANEL_HALFS * 2;          // 14336
constexpr int C1_OFF  = H2A_OFF + HPANEL_HALFS * 2;          // 28672
constexpr int C2_OFF  = C1_OFF + NNT * 2 * 4 * 2 * 8 * 4;    // +25600 = 54272
constexpr int WL_OFF  = C2_OFF + NNT * 2 * 4 * 2 * 8 * 4;    // +25600 = 79872
constexpr int SMEM_BYTES = WL_OFF + HC * 4;                  // 80672

#if defined(__HIP_DEVICE_COMPILE__) && __has_builtin(__builtin_amdgcn_tanhf)
__device__ __forceinline__ float tanh_(float x) { return __builtin_amdgcn_tanhf(x); }
#else
__device__ __forceinline__ float tanh_(float x) { return tanhf(x); }
#endif
__device__ __forceinline__ float sigmoid_(float x) { return 0.5f * tanh_(0.5f * x) + 0.5f; }

// ---------------------------------------------------------------------------
// Pack weights into WMMA-ready f16 B-panels (gate-interleaved, bias/x folded).
// Panel element order: ((nt*NKC + kc)*32 + lane)*16 + j
// B striping (ISA 7.12.2): n = lane%16 ; k = kc*32 + (lane>=16?8:0) + (j&7) + (j>=8?16:0)
// ---------------------------------------------------------------------------
__global__ void lstm_pack_weights(
    const float* __restrict__ W_ih1, const float* __restrict__ W_hh1,
    const float* __restrict__ b_ih1, const float* __restrict__ b_hh1,
    const float* __restrict__ W_ih2, const float* __restrict__ W_hh2,
    const float* __restrict__ b_ih2, const float* __restrict__ b_hh2,
    _Float16* __restrict__ P)
{
  int idx  = blockIdx.x * blockDim.x + threadIdx.x;
  int j    = idx & 15;
  int lane = (idx >> 4) & 31;
  int rest = idx >> 9;
  int kc   = rest % NKC;  rest /= NKC;
  int nt   = rest % NNT;
  int panel= rest / NNT;
  if (panel >= 3) return;

  int n = nt * 16 + (lane & 15);
  int k = kc * 32 + ((lane >> 4) & 1) * 8 + (j & 7) + (j >> 3) * 16;
  int u = n >> 2, gate = n & 3;
  int jw = gate * HC + u;                 // original PyTorch gate-row (i,f,g,o)

  float v = 0.0f;
  if (panel == 0) {                       // layer1: W_hh1 | x-col | bias-col
    if (k < HC)            v = W_hh1[jw * HC + k];
    else if (k == HC)      v = W_ih1[jw];
    else if (k == HC + 1)  v = b_ih1[jw] + b_hh1[jw];
  } else if (panel == 1) {                // layer2 input path
    if (k < HC)            v = W_ih2[jw * HC + k];
    else if (k == HC + 1)  v = b_ih2[jw] + b_hh2[jw];
  } else {                                // layer2 recurrent path
    if (k < HC)            v = W_hh2[jw * HC + k];
  }
  P[idx] = (_Float16)v;
}

// ---------------------------------------------------------------------------
// Fragment helpers. h-state lives in LDS already in A-fragment striping:
// slot(tile(mt,kc), lane, elem) -> h[b][u] with
//   b = mt*16 + (lane&15),  u = kc*32 + ((lane>=16)?8:0) + (elem&7) + (elem>=8?16:0)
// ---------------------------------------------------------------------------
__device__ __forceinline__ v16h load_a(const _Float16* hA, int mt, int kc, int lane) {
  return *(const v16h*)(hA + (((mt * NKC + kc) * 32 + lane) << 4));
}

// B fragment: Bl = panel + wave*NKC*512 + lane*16 (uniform SGPR base + lane voffset),
// per-fragment byte offset (i*NWAVE*NKC + kc)*1024 is a compile-time immediate.
__device__ __forceinline__ v8f mm_tile(const _Float16* hA, const _Float16* Bl,
                                       int i, int mt, int lane, v8f acc) {
#pragma unroll
  for (int kc = 0; kc < NKC; ++kc) {
    v16h a = load_a(hA, mt, kc, lane);
    v16h b = *(const v16h*)(Bl + ((i * NWAVE * NKC + kc) << 9));
    acc = __builtin_amdgcn_wmma_f32_16x16x32_f16(false, a, false, b, (short)0, acc,
                                                 false, false);
  }
  return acc;
}

// LSTM pointwise update. D striping: elem v -> row m = v + (lane>=16?8:0),
// col n = lane%16; gate-interleaved cols: lanes {base..base+3} = {i,f,g,o}.
__device__ __forceinline__ void cell_update(v8f g, float* cl, _Float16* hA,
                                            int nt, int mt, int lane) {
  const int ug      = (lane & 15) >> 2;
  const int mhalf   = (lane >> 4) & 1;
  const int srcBase = (lane & 16) | (lane & 12);
  float* cptr = cl + ((((nt * 2 + mt) * 4 + ug) * 2 + mhalf) << 3);
  v8f cold = *(const v8f*)cptr;
  v8f cnew;

  const int u    = nt * 4 + ug;
  const int kc   = u >> 5, krel = u & 31;
  const int elem = (krel & 7) + ((krel >> 4) & 1) * 8;
  const int lbit = ((krel >> 3) & 1) * 16;
  _Float16* hbase = hA + (((mt * NKC + kc) * 32 + mhalf * 8 + lbit) << 4) + elem;

#pragma unroll
  for (int v = 0; v < 8; ++v) {
    float gi = __shfl(g[v], srcBase + 0, 32);
    float gf = __shfl(g[v], srcBase + 1, 32);
    float gg = __shfl(g[v], srcBase + 2, 32);
    float go = __shfl(g[v], srcBase + 3, 32);
    float cn = sigmoid_(gf) * cold[v] + sigmoid_(gi) * tanh_(gg);
    cnew[v] = cn;
    float hn = sigmoid_(go) * tanh_(cn);
    if ((lane & 3) == 0) hbase[v << 4] = (_Float16)hn;   // lane_t = mhalf*8+v+lbit
  }
  if ((lane & 3) == 0) *(v8f*)cptr = cnew;
}

// ---------------------------------------------------------------------------
// Persistent single-workgroup LSTM: 16 wave32, weights streamed from L2,
// h-state (f16, fragment layout) + c-state (f32) in LDS.
// ---------------------------------------------------------------------------
__global__ void __launch_bounds__(NTHREADS, 1)
lstm_persistent(const float* __restrict__ x,
                const _Float16* __restrict__ P1g,
                const _Float16* __restrict__ P2ig,
                const _Float16* __restrict__ P2hg,
                const float* __restrict__ W_l,
                const float* __restrict__ b_l,
                float* __restrict__ out, int T)
{
  extern __shared__ __align__(32) char smem[];
  _Float16* h1A = (_Float16*)(smem + H1A_OFF);
  _Float16* h2A = (_Float16*)(smem + H2A_OFF);
  float*    c1l = (float*)(smem + C1_OFF);
  float*    c2l = (float*)(smem + C2_OFF);
  float*    wls = (float*)(smem + WL_OFF);

  const int tid  = threadIdx.x;
  const int wave = tid >> 5;
  const int lane = tid & 31;

  for (int i = tid; i < HPANEL_HALFS; i += NTHREADS) {
    h1A[i] = (_Float16)0.0f; h2A[i] = (_Float16)0.0f;
  }
  for (int i = tid; i < 2 * NNT * 2 * 4 * 2 * 8; i += NTHREADS) c1l[i] = 0.0f; // c1+c2
  for (int i = tid; i < HC; i += NTHREADS) wls[i] = W_l[i];
  __syncthreads();
  if (tid < BATCH) {       // bias-one column: u=201 -> kc=6,krel=9 -> lane+16, elem 1
    int mt = tid >> 4;
    h1A[(((mt * NKC + 6) * 32 + (tid & 15) + 16) << 4) + 1] = (_Float16)1.0f;
  }
  __syncthreads();

  const int myCnt = (wave < 2) ? 4 : 3;       // tiles nt = wave + 16*i (< 50)
  const int waveLane = wave * (NKC * 512) + (lane << 4);   // halfs into each panel
  const v8f vzero = {0.f,0.f,0.f,0.f,0.f,0.f,0.f,0.f};

  for (int t = 0; t < T; ++t) {
    // Launder a 32-bit zero through an SGPR each iteration: the weight loads
    // stay global (pointer provenance intact) but their values cannot be
    // proven loop-invariant, so nothing is hoisted/spilled across t.
    unsigned lz = 0;
    asm volatile("" : "+s"(lz));
    const _Float16* B1  = P1g  + lz + waveLane;
    const _Float16* B2i = P2ig + lz + waveLane;
    const _Float16* B2h = P2hg + lz + waveLane;

    if (tid < BATCH) {     // x_t column: u=200 -> kc=6,krel=8 -> lane+16, elem 0
      int mt = tid >> 4;
      h1A[((mt * NKC + 6) * 32 + (tid & 15) + 16) << 4] =
          (_Float16)x[(size_t)tid * T + t];
    }
    __syncthreads();

    v8f acc[NTMAX][2];
    // ---- layer 1 gates: [h1 | x_t | 1] @ P1 -----------------------------
#pragma unroll
    for (int i = 0; i < NTMAX; ++i) if (i < myCnt) {
#pragma unroll
      for (int mt = 0; mt < 2; ++mt)
        acc[i][mt] = mm_tile(h1A, B1, i, mt, lane, vzero);
    }
    __syncthreads();                          // all h1A reads done
#pragma unroll
    for (int i = 0; i < NTMAX; ++i) if (i < myCnt) {
      int nt = wave + NWAVE * i;
#pragma unroll
      for (int mt = 0; mt < 2; ++mt)
        cell_update(acc[i][mt], c1l, h1A, nt, mt, lane);
    }
    __syncthreads();                          // h1(t) ready
    // ---- layer 2 gates: [h1|0|1]@P2i + [h2|0|0]@P2h ---------------------
#pragma unroll
    for (int i = 0; i < NTMAX; ++i) if (i < myCnt) {
#pragma unroll
      for (int mt = 0; mt < 2; ++mt) {
        v8f a2 = mm_tile(h1A, B2i, i, mt, lane, vzero);
        acc[i][mt] = mm_tile(h2A, B2h, i, mt, lane, a2);
      }
    }
    __syncthreads();                          // all h2A reads done
#pragma unroll
    for (int i = 0; i < NTMAX; ++i) if (i < myCnt) {
      int nt = wave + NWAVE * i;
#pragma unroll
      for (int mt = 0; mt < 2; ++mt)
        cell_update(acc[i][mt], c2l, h2A, nt, mt, lane);
    }
    __syncthreads();                          // h2(t) ready
    // ---- output projection (wave 0; others stall at next barrier) -------
    if (tid < BATCH) {
      int mt = tid >> 4;
      float s = b_l[0];
#pragma unroll
      for (int cc = 0; cc < 25; ++cc) {       // u = kc*32 + sub*8 + q
        int kc = cc >> 2, sub = cc & 3;
        const v8h hv = *(const v8h*)(h2A +
            (((mt * NKC + kc) * 32 + (tid & 15) + (sub & 1) * 16) << 4) +
            (sub >> 1) * 8);
#pragma unroll
        for (int q = 0; q < 8; ++q) s += wls[kc * 32 + sub * 8 + q] * (float)hv[q];
      }
      out[(size_t)tid * T + t] = s;
    }
  }
}

// ---------------------------------------------------------------------------
extern "C" void kernel_launch(void* const* d_in, const int* in_sizes, int n_in,
                              void* d_out, int out_size, void* d_ws, size_t ws_size,
                              hipStream_t stream) {
  const float* x     = (const float*)d_in[0];
  const float* W_ih1 = (const float*)d_in[1];
  const float* W_hh1 = (const float*)d_in[2];
  const float* b_ih1 = (const float*)d_in[3];
  const float* b_hh1 = (const float*)d_in[4];
  const float* W_ih2 = (const float*)d_in[5];
  const float* W_hh2 = (const float*)d_in[6];
  const float* b_ih2 = (const float*)d_in[7];
  const float* b_hh2 = (const float*)d_in[8];
  const float* W_l   = (const float*)d_in[9];
  const float* b_l   = (const float*)d_in[10];

  const int T = in_sizes[0] / BATCH;                    // 8192

  _Float16* P = (_Float16*)d_ws;                        // 3 panels: 1,075,200 B

  const int packElems = 3 * PANEL_HALFS;                // 537600
  lstm_pack_weights<<<packElems / 256, 256, 0, stream>>>(
      W_ih1, W_hh1, b_ih1, b_hh1, W_ih2, W_hh2, b_ih2, b_hh2, P);

  (void)hipFuncSetAttribute((const void*)lstm_persistent,
                            hipFuncAttributeMaxDynamicSharedMemorySize, SMEM_BYTES);
  lstm_persistent<<<1, NTHREADS, SMEM_BYTES, stream>>>(
      x, P, P + PANEL_HALFS, P + 2 * PANEL_HALFS,
      W_l, b_l, (float*)d_out, T);
}
